// Critic_82128364634640
// MI455X (gfx1250) — matrix operand order repaired and checked by reading, not compiled, not run on hardware
//
#include <hip/hip_runtime.h>
#include <hip/hip_fp16.h>
#include <math.h>

typedef __attribute__((ext_vector_type(16))) _Float16 v16h;
typedef __attribute__((ext_vector_type(8)))  float    v8f;
typedef __attribute__((ext_vector_type(4)))  unsigned int v4u;
typedef __attribute__((ext_vector_type(8)))  int      v8i;
typedef __attribute__((ext_vector_type(4)))  int      v4i;

#define SEQ   64
#define BGRP  64
#define NPG   256
#define NTOT  (BGRP*NPG)
#define EDIM  64
#define HDIM  64
#define FDIM  67      // E + 3 extra columns
#define XH_LD 136     // halves: 128 ([x|h]) + 8 pad  -> 272 B row (16B aligned)
#define WC_LD 264     // halves: 256 + 8 pad          -> 528 B row (16B aligned)
#define W7_LD 72      // halves: 64 + 8 pad           -> 144 B row (16B aligned)
#define XTILE (NPG*EDIM)   // 16384 f32 per staged x tile

__device__ __forceinline__ float sigm(float x)  { return 1.0f / (1.0f + __expf(-x)); }
__device__ __forceinline__ float tanh_f(float x){ float e = __expf(2.0f*x); return 1.0f - 2.0f/(e + 1.0f); }

// A-fragment (16x32 f16, M x K): lane L -> row M = row0 + (L&15); half = L>>4.
// VGPR0..3 hold K = half*8 + 0..7 ; VGPR4..7 hold K = 16 + half*8 + 0..7 (ISA 7.12.2)
__device__ __forceinline__ v16h load_frag_a(const _Float16* base, int ldh,
                                            int row0, int k0, int lane) {
  const int m = row0 + (lane & 15);
  const int half = lane >> 4;
  const _Float16* p = base + m * ldh + k0 + half * 8;
  union { v16h v; uint4 u[2]; } f;
  f.u[0] = *(const uint4*)(p);        // K = half*8 + 0..7
  f.u[1] = *(const uint4*)(p + 16);   // K = 16 + half*8 + 0..7
  return f.v;
}

// B-fragment (32x16 f16, K x N): lane L -> col N = n0 + (L&15); half = L>>4;
// elems 0..7 hold K = half*8 + 0..7 ; elems 8..15 hold K = 16 + half*8 + 0..7.
__device__ __forceinline__ v16h load_frag_b(const _Float16* base, int ldh,
                                            int k0, int n0, int lane) {
  const _Float16* p = base + (k0 + (lane >> 4) * 8) * ldh + n0 + (lane & 15);
  v16h v;
  #pragma unroll
  for (int s = 0; s < 8; ++s) { v[s] = p[s * ldh]; v[8 + s] = p[(16 + s) * ldh]; }
  return v;
}

// TDM: async-DMA a 2D tile (64 f32 x 256 rows, row stride 67 f32) from global
// into contiguous LDS. D# bitfields per CDNA5 ISA 8.3/8.4. Tracked by TENSORcnt.
// This toolchain declares the 6-arg builtin (extra int32x8 group, zero-filled).
__device__ __forceinline__ void tdm_load_x_tile(const float* gsrc, unsigned lds_off) {
  unsigned long long ga = (unsigned long long)(uintptr_t)gsrc;
  union { v4u v; unsigned u[4]; } g0;
  g0.u[0] = 1u;                                   // count=1 (valid), user descriptor
  g0.u[1] = lds_off;                              // lds_addr (bytes, WG-relative)
  g0.u[2] = (unsigned)(ga & 0xFFFFFFFFu);         // global_addr[31:0]
  g0.u[3] = (unsigned)((ga >> 32) & 0x01FFFFFFu)  // global_addr[56:32]
            | (2u << 30);                         // type = 2 ("image")
  union { v8i v; int i[8]; } g1;
  g1.i[0] = (2 << 16);                            // data_size = 2 -> 4 bytes
  g1.i[1] = (FDIM & 0xFFFF) << 16;                // tensor_dim0[15:0] = 67
  g1.i[2] = (NPG & 0xFFFF) << 16;                 // tensor_dim1[15:0] = 256
  g1.i[3] = (EDIM & 0xFFFF) << 16;                // tile_dim0 = 64
  g1.i[4] = (NPG & 0xFFFF);                       // tile_dim1 = 256, tile_dim2 = 0
  g1.i[5] = FDIM;                                 // tensor_dim0_stride[31:0] = 67
  g1.i[6] = 0;                                    // stride hi / dim1_stride lo
  g1.i[7] = 0;
  v4i z4 = {0, 0, 0, 0};                          // groups 2/3: dims>2 unused
  v8i z8 = {0, 0, 0, 0, 0, 0, 0, 0};              // extra descriptor group: unused
  __builtin_amdgcn_tensor_load_to_lds(g0.v, g1.v, z4, z4, z8, 0);
}

__global__ __launch_bounds__(512)
void critic_fused(const float* __restrict__ feat,
                  const float* __restrict__ W_ih, const float* __restrict__ W_hh,
                  const float* __restrict__ b_ih, const float* __restrict__ b_hh,
                  const float* __restrict__ W6,  const float* __restrict__ b6,
                  const float* __restrict__ W7,  const float* __restrict__ b7,
                  const float* __restrict__ W5,  const float* __restrict__ b5,
                  float* __restrict__ out)
{
  // ~297 KB static LDS (WGP allows up to 320 KB)
  __shared__ __align__(16) float    stg[2 * XTILE];       // TDM double buffer (f32 x tiles)
  __shared__ __align__(16) _Float16 xh[NPG * XH_LD];      // [x(64) | h(64)] per node, f16
  __shared__ __align__(16) _Float16 Wcat[128 * WC_LD];    // [W_ih;W_hh]^T : (k, gate-col)
  __shared__ __align__(16) _Float16 W7t[64 * W7_LD];      // W7^T : (k, out-col)
  __shared__ float W6s[64 * 64];
  __shared__ float bsumS[256];
  __shared__ float reachS[NPG];
  __shared__ float b7S[64], b6S[64], w5S[128];
  __shared__ float poolAcc[64];
  __shared__ float gsqS[64];
  __shared__ float wavemaxS[16];
  __shared__ float qgS, b5S;

  const int tid  = threadIdx.x;
  const int lane = tid & 31;
  const int wave = tid >> 5;        // 16 waves, each owns 16 node-rows
  const int half = lane >> 4;
  const int c16  = lane & 15;
  const int grp  = blockIdx.x;      // one batch group (256 nodes) per workgroup
  const int rowBase = wave * 16;

  const unsigned stgOff0 = (unsigned)(uintptr_t)(void*)&stg[0];      // LDS byte offsets
  const unsigned stgOff1 = (unsigned)(uintptr_t)(void*)&stg[XTILE];

  // ---- one-time preload of weights / biases / reach mask, zero h ----
  for (int idx = tid; idx < 128 * 256; idx += 512) {
    int k = idx >> 8, n = idx & 255;
    float w = (k < 64) ? W_ih[n * 64 + k] : W_hh[n * 64 + (k - 64)];
    Wcat[k * WC_LD + n] = (_Float16)w;
  }
  for (int idx = tid; idx < 64 * 64; idx += 512) {
    int k = idx >> 6, j = idx & 63;
    W7t[k * W7_LD + j] = (_Float16)W7[j * 64 + k];
    W6s[idx] = W6[idx];
  }
  for (int idx = tid; idx < 256; idx += 512) {
    bsumS[idx]  = b_ih[idx] + b_hh[idx];
    reachS[idx] = feat[((size_t)grp * NPG + idx) * FDIM + (EDIM + 1)];
  }
  if (tid < 64)  { b7S[tid] = b7[tid]; b6S[tid] = b6[tid]; }
  if (tid < 128) { w5S[tid] = W5[tid]; }
  if (tid == 0)  { b5S = b5[0]; }
  for (int idx = tid; idx < NPG * 64; idx += 512) {
    int m = idx >> 6, j = idx & 63;
    xh[m * XH_LD + 64 + j] = (_Float16)0.0f;
  }

  // prologue: kick off DMA of x_0 (TENSORcnt = 1 on wave 0)
  if (wave == 0) {
    tdm_load_x_tile(feat + (size_t)grp * NPG * FDIM, stgOff0);
  }

  // per-lane cell state: (m = rowBase + 8*half + r , j = 16*tt + c16)
  float cst[4][8];
  #pragma unroll
  for (int a = 0; a < 4; ++a)
    #pragma unroll
    for (int r = 0; r < 8; ++r) cst[a][r] = 0.0f;

  v8f zero8;
  #pragma unroll
  for (int i = 0; i < 8; ++i) zero8[i] = 0.0f;

  for (int t = 0; t < SEQ; ++t) {
    // ---- stage 0: issue DMA for x_{t+1}, then wait for x_t (in-order TENSORcnt) ----
    if (wave == 0) {
      if (t + 1 < SEQ) {
        const float* nx = feat + ((size_t)(t + 1) * NTOT + (size_t)grp * NPG) * FDIM;
        tdm_load_x_tile(nx, ((t + 1) & 1) ? stgOff1 : stgOff0);
        __builtin_amdgcn_s_wait_tensorcnt((short)1);   // x_t landed; x_{t+1} in flight
      } else {
        __builtin_amdgcn_s_wait_tensorcnt((short)0);
      }
    }
    __syncthreads();   // publish DMA'd tile to all waves

    // ---- stage 1: convert staged f32 tile -> f16 x-part of xh; zero pool acc ----
    if (tid < 64) poolAcc[tid] = 0.0f;
    {
      const float4* src4 = (const float4*)&stg[(t & 1) * XTILE];
      for (int idx = tid; idx < (NPG * EDIM) / 4; idx += 512) {
        int m  = idx >> 4;            // 16 float4 per 64-wide row
        int e4 = (idx & 15) * 4;
        float4 xv = src4[idx];
        union { _Float16 h[4]; unsigned long long u; } pk;
        pk.h[0] = (_Float16)xv.x; pk.h[1] = (_Float16)xv.y;
        pk.h[2] = (_Float16)xv.z; pk.h[3] = (_Float16)xv.w;
        *(unsigned long long*)&xh[m * XH_LD + e4] = pk.u;
      }
    }
    __syncthreads();

    // ---- stage 2: gates = [x|h] (256x128) @ Wcat (128x256), f32 accum ----
    v8f accg[16];
    #pragma unroll
    for (int tt = 0; tt < 16; ++tt) accg[tt] = zero8;
    #pragma unroll
    for (int kb = 0; kb < 4; ++kb) {
      v16h af = load_frag_a(xh, XH_LD, rowBase, 32 * kb, lane);
      #pragma unroll
      for (int tt = 0; tt < 16; ++tt) {
        v16h bf = load_frag_b(Wcat, WC_LD, 32 * kb, 16 * tt, lane);
        accg[tt] = __builtin_amdgcn_wmma_f32_16x16x32_f16(
            false, af, false, bf, (short)0, accg[tt], false, false);
      }
    }

    // ---- LSTM elementwise: i,f,g,o are in same lane/reg across tile quads.
    //      Also fold this lane's 8-row h-sum into the pooling accumulator. ----
    #pragma unroll
    for (int tt = 0; tt < 4; ++tt) {
      const int j  = 16 * tt + c16;
      const float bi = bsumS[j], bff = bsumS[64 + j], bg = bsumS[128 + j], bo = bsumS[192 + j];
      float hsum = 0.0f;
      #pragma unroll
      for (int r = 0; r < 8; ++r) {
        float ig = sigm(accg[tt][r] + bi);
        float fg = sigm(accg[tt + 4][r] + bff);
        float gg = tanh_f(accg[tt + 8][r] + bg);
        float og = sigm(accg[tt + 12][r] + bo);
        float cn = fg * cst[tt][r] + ig * gg;
        cst[tt][r] = cn;
        float hn = og * tanh_f(cn);
        hsum += hn;
        int m = rowBase + 8 * half + r;
        xh[m * XH_LD + 64 + j] = (_Float16)hn;
      }
      atomicAdd(&poolAcc[j], hsum);   // ds_add_f32: 32 partials per column
    }
    __syncthreads();   // h + column sums complete for all 256 nodes

    // ---- stage 3: local_action = h @ W7^T (K=64), then global head ----
    v8f accl[4];
    #pragma unroll
    for (int tt = 0; tt < 4; ++tt) accl[tt] = zero8;
    #pragma unroll
    for (int kb = 0; kb < 2; ++kb) {
      v16h af = load_frag_a(xh, XH_LD, rowBase, 64 + 32 * kb, lane);
      #pragma unroll
      for (int tt = 0; tt < 4; ++tt) {
        v16h bf = load_frag_b(W7t, W7_LD, 32 * kb, 16 * tt, lane);
        accl[tt] = __builtin_amdgcn_wmma_f32_16x16x32_f16(
            false, af, false, bf, (short)0, accl[tt], false, false);
      }
    }
    if (tid < 64) {   // gs = mean(h) @ W6^T + b6 ; relu ; * w5[:64]
      float g = b6S[tid];
      const float* wrow = &W6s[tid * 64];
      for (int j = 0; j < 64; ++j) g += (poolAcc[j] * (1.0f / 256.0f)) * wrow[j];
      g = g > 0.0f ? g : 0.0f;
      gsqS[tid] = g * w5S[tid];
    }
    __syncthreads();
    if (tid == 0) {
      float s = b5S;
      for (int j = 0; j < 64; ++j) s += gsqS[j];
      qgS = s;
    }
    __syncthreads();

    // ---- stage 4: per-node q = qg + relu(la + b7) . w5[64:], mask, group max ----
    const float qg = qgS;
    float mxh = -INFINITY;
    #pragma unroll
    for (int r = 0; r < 8; ++r) {
      float p = 0.0f;
      #pragma unroll
      for (int tt = 0; tt < 4; ++tt) {
        int j = 16 * tt + c16;
        float v = accl[tt][r] + b7S[j];
        v = v > 0.0f ? v : 0.0f;
        p += v * w5S[64 + j];
      }
      #pragma unroll
      for (int off = 1; off < 16; off <<= 1)   // reduce over the 16 lanes of a half
        p += __shfl_xor(p, off, 32);
      int m = rowBase + 8 * half + r;
      float q = qg + p;
      q = (reachS[m] > 0.5f) ? q : -INFINITY;
      mxh = fmaxf(mxh, q);
    }
    mxh = fmaxf(mxh, __shfl_xor(mxh, 16, 32));  // combine both halves
    if (lane == 0) wavemaxS[wave] = mxh;
    __syncthreads();

    if (tid == 0) {
      float v = -INFINITY;
      #pragma unroll
      for (int w = 0; w < 16; ++w) v = fmaxf(v, wavemaxS[w]);
      out[t * BGRP + grp] = v;
    }
    // next iteration's stage-0 barrier protects the xh x-region overwrite
  }
}

extern "C" void kernel_launch(void* const* d_in, const int* in_sizes, int n_in,
                              void* d_out, int out_size, void* d_ws, size_t ws_size,
                              hipStream_t stream) {
  (void)in_sizes; (void)n_in; (void)d_ws; (void)ws_size; (void)out_size;
  const float* feat = (const float*)d_in[0];
  const float* W_ih = (const float*)d_in[1];
  const float* W_hh = (const float*)d_in[2];
  const float* b_ih = (const float*)d_in[3];
  const float* b_hh = (const float*)d_in[4];
  const float* W6   = (const float*)d_in[5];
  const float* b6   = (const float*)d_in[6];
  const float* W7   = (const float*)d_in[7];
  const float* b7   = (const float*)d_in[8];
  const float* W5   = (const float*)d_in[9];
  const float* b5   = (const float*)d_in[10];
  float* out = (float*)d_out;

  critic_fused<<<dim3(BGRP), dim3(512), 0, stream>>>(
      feat, W_ih, W_hh, b_ih, b_hh, W6, b6, W7, b7, W5, b5, out);
}
